// BinaryConv2d_27882927686416
// MI455X (gfx1250) — 8-bit matrix operand order repaired and checked by reading, compiled, not run on hardware
//
#include <hip/hip_runtime.h>
#include <stdint.h>

#define BB 16
#define CC 256
#define HH 64
#define WW 64

#define YS (66 * 256)
#define BS (66 * YS)
#define ACT_BYTES (16 * BS)

#define WS_OFF   ACT_BYTES
#define WS_BYTES (256 * 2304)
#define SC_OFF   (WS_OFF + WS_BYTES)

typedef int v8i __attribute__((ext_vector_type(8)));

__device__ __forceinline__ int8_t sign8(float v) {
    return v > 0.f ? (int8_t)1 : (v < 0.f ? (int8_t)-1 : (int8_t)0);
}

__global__ void k_zero(uint4* __restrict__ p, int n) {
    int i = blockIdx.x * blockDim.x + threadIdx.x;
    if (i < n) p[i] = make_uint4(0u, 0u, 0u, 0u);
}

__global__ void k_binact(const float* __restrict__ x,
                         const float* __restrict__ bias,
                         int8_t* __restrict__ actS) {
    __shared__ int8_t tile[64][68];
    const int c0 = blockIdx.x * 64;
    const int y  = blockIdx.y;
    const int b  = blockIdx.z;
    const int lw = threadIdx.x & 63;
    const int lq = threadIdx.x >> 6;

#pragma unroll
    for (int co = 0; co < 16; ++co) {
        int c = co * 4 + lq;
        float v = x[(size_t)((b * CC + c0 + c) * 4096) + y * 64 + lw] + bias[c0 + c];
        tile[c][lw] = sign8(v);
    }
    __syncthreads();

    int8_t* dst = actS + (size_t)b * BS + (size_t)(y + 1) * YS + 256 + (c0 + lw);
#pragma unroll
    for (int wo = 0; wo < 16; ++wo) {
        int w = wo * 4 + lq;
        dst[(size_t)w * 256] = tile[lw][w];
    }
}

__global__ void k_binw(const float* __restrict__ w,
                       int8_t* __restrict__ wS,
                       float* __restrict__ scale) {
    __shared__ float red[256];
    const int o  = blockIdx.x;
    const int ci = threadIdx.x;
    float s = 0.f;
#pragma unroll
    for (int tap = 0; tap < 9; ++tap) {
        float v = w[(size_t)(o * 256 + ci) * 9 + tap];
        s += fabsf(v);
        int kg    = tap * 256 + ci;
        int kstep = kg >> 6;
        int k     = kg & 63;
        int q     = k >> 4;
        int rem   = k & 15;
        int hf    = rem >> 3;
        int off   = rem & 7;
        wS[(size_t)o * 2304 + kstep * 64 + (q & 1) * 32 + (q >> 1) * 16 + rem] = sign8(v);
    }
    red[ci] = s;
    __syncthreads();
    for (int offr = 128; offr > 0; offr >>= 1) {
        if (ci < offr) red[ci] += red[ci + offr];
        __syncthreads();
    }
    if (ci == 0) scale[o] = red[0] / 2304.0f;
}

__device__ __forceinline__ v8i pack8(uint4 lo, uint4 hi) {
    v8i r;
    r[0] = (int)lo.x; r[1] = (int)lo.y; r[2] = (int)lo.z; r[3] = (int)lo.w;
    r[4] = (int)hi.x; r[5] = (int)hi.y; r[6] = (int)hi.z; r[7] = (int)hi.w;
    return r;
}
__device__ __forceinline__ v8i loadA(const int8_t* q) {
    return pack8(*(const uint4*)(q), *(const uint4*)(q + 16));
}
__device__ __forceinline__ v8i loadB(const int8_t* p) {
    return pack8(*(const uint4*)(p), *(const uint4*)(p + 32));
}

__global__ void __launch_bounds__(256)
k_bgemm(const int8_t* __restrict__ actS, const int8_t* __restrict__ wS,
        const float* __restrict__ scale, const float* __restrict__ pb0,
        const float* __restrict__ alpha, const float* __restrict__ pb1,
        const float* __restrict__ xin, float* __restrict__ out) {
    const int tid  = threadIdx.x;
    const int lane = tid & 31;
    const int col  = lane & 15;
    const int half = lane >> 4;
    const int wv   = tid >> 5;
    const int m_w  = blockIdx.y * 64 + (wv >> 2) * 32;
    const int n_w  = blockIdx.x * 256 + (wv & 3) * 64;

    const int b  = n_w >> 12;
    const int h  = (n_w >> 6) & 63;
    const int wb = n_w & 63;

    v8i acc[2][4] = {};

    const int8_t* aBase0 = wS + (size_t)(m_w + col) * 2304 + half * 32;
    const int8_t* aBase1 = aBase0 + (size_t)16 * 2304;
    const int8_t* actB   = actS + (size_t)b * BS + half * 16;

    for (int kh = 0; kh < 3; ++kh) {
        const int8_t* rowp = actB + (size_t)(h + kh) * YS;
        __builtin_prefetch(rowp + YS, 0, 1);
        for (int kw = 0; kw < 3; ++kw) {
            const int8_t* p0 = rowp + (size_t)(wb + col + kw) * 256;
            const int8_t* p1 = p0 + 16 * 256;
            const int8_t* p2 = p0 + 32 * 256;
            const int8_t* p3 = p0 + 48 * 256;
            const int kstep0 = (kh * 3 + kw) << 2;
#pragma unroll
            for (int kq = 0; kq < 4; ++kq) {
                const int cb = kq << 6;
                v8i bf0 = loadB(p0 + cb);
                v8i bf1 = loadB(p1 + cb);
                v8i bf2 = loadB(p2 + cb);
                v8i bf3 = loadB(p3 + cb);
                const size_t ko = (size_t)(kstep0 + kq) * 64;
                v8i af0 = loadA(aBase0 + ko);
                v8i af1 = loadA(aBase1 + ko);
                acc[0][0] = __builtin_amdgcn_wmma_i32_16x16x64_iu8(
                    true, af0, true, bf0, acc[0][0], false, false);
                acc[0][1] = __builtin_amdgcn_wmma_i32_16x16x64_iu8(
                    true, af0, true, bf1, acc[0][1], false, false);
                acc[0][2] = __builtin_amdgcn_wmma_i32_16x16x64_iu8(
                    true, af0, true, bf2, acc[0][2], false, false);
                acc[0][3] = __builtin_amdgcn_wmma_i32_16x16x64_iu8(
                    true, af0, true, bf3, acc[0][3], false, false);
                acc[1][0] = __builtin_amdgcn_wmma_i32_16x16x64_iu8(
                    true, af1, true, bf0, acc[1][0], false, false);
                acc[1][1] = __builtin_amdgcn_wmma_i32_16x16x64_iu8(
                    true, af1, true, bf1, acc[1][1], false, false);
                acc[1][2] = __builtin_amdgcn_wmma_i32_16x16x64_iu8(
                    true, af1, true, bf2, acc[1][2], false, false);
                acc[1][3] = __builtin_amdgcn_wmma_i32_16x16x64_iu8(
                    true, af1, true, bf3, acc[1][3], false, false);
            }
        }
    }

#pragma unroll
    for (int mt = 0; mt < 2; ++mt) {
#pragma unroll
        for (int nt = 0; nt < 4; ++nt) {
#pragma unroll
            for (int r = 0; r < 8; ++r) {
                int m  = m_w + mt * 16 + r + half * 8;
                int wp = wb + nt * 16 + col;
                size_t idx = ((size_t)(b * 256 + m) << 12) + (h << 6) + wp;
                float v = scale[m] * (float)acc[mt][nt][r] + pb0[m];
                v = v >= 0.f ? v : alpha[m] * v;
                v += pb1[m] + xin[idx];
                out[idx] = v;
            }
        }
    }
}

extern "C" void kernel_launch(void* const* d_in, const int* in_sizes, int n_in,
                              void* d_out, int out_size, void* d_ws, size_t ws_size,
                              hipStream_t stream) {
    const float* x     = (const float*)d_in[0];
    const float* mbias = (const float*)d_in[1];
    const float* cw    = (const float*)d_in[2];
    const float* pb0   = (const float*)d_in[3];
    const float* alpha = (const float*)d_in[4];
    const float* pb1   = (const float*)d_in[5];
    float* out = (float*)d_out;

    uint8_t* ws   = (uint8_t*)d_ws;
    int8_t*  actS = (int8_t*)ws;
    int8_t*  wS   = (int8_t*)(ws + WS_OFF);
    float*   scl  = (float*)(ws + SC_OFF);

    const int nz = ACT_BYTES / 16;
    k_zero<<<(nz + 255) / 256, 256, 0, stream>>>((uint4*)actS, nz);
    k_binact<<<dim3(4, 64, 16), 256, 0, stream>>>(x, mbias, actS);
    k_binw<<<256, 256, 0, stream>>>(cw, wS, scl);
    k_bgemm<<<dim3(256, 4), 256, 0, stream>>>(actS, wS, scl, pb0, alpha, pb1, x, out);
}
